// ZetaLSTMResonant_26259430047901
// MI455X (gfx1250) — matrix operand order repaired and checked by reading, not compiled, not run on hardware
//
#include <hip/hip_runtime.h>
#include <hip/hip_bf16.h>
#include <math.h>

// Problem dims
#define B_    256
#define S_    512
#define I_    256
#define H_    512
#define G4    2048          // 4*H
#define MM    15
#define WIN_  5
#define KPACK 768           // H + I  (fused [h|x] K dimension)
#define NWG   128           // persistent workgroups (16 batch tiles x 8 hidden chunks)
#define TPB   128           // 4 waves per workgroup
#define JCH   64            // hidden columns per chunk (4 waves x 16)

typedef __attribute__((ext_vector_type(16))) __bf16 v16bf;
typedef __attribute__((ext_vector_type(8)))  float  v8f;
typedef __attribute__((ext_vector_type(4)))  unsigned v4u;
typedef __attribute__((ext_vector_type(8)))  int   v8i;
typedef __attribute__((ext_vector_type(4)))  int   v4i;

#if __has_builtin(__builtin_amdgcn_tensor_load_to_lds)
#define HAVE_TDM 1
#if defined(__clang_major__) && __clang_major__ >= 23
#define TDM_LOAD(g0, g1, g2, g3) \
  __builtin_amdgcn_tensor_load_to_lds(g0, g1, g2, g3, (v8i){0,0,0,0,0,0,0,0}, 0)
#else
#define TDM_LOAD(g0, g1, g2, g3) \
  __builtin_amdgcn_tensor_load_to_lds(g0, g1, g2, g3, 0)
#endif
#else
#define HAVE_TDM 0
#endif

__constant__ float c_gam[MM] = {
  14.134725f, 21.02204f, 25.010858f, 30.424876f, 32.935062f,
  37.586178f, 40.91872f, 43.327073f, 48.005151f, 49.773832f,
  52.970321f, 56.446248f, 59.347044f, 60.831779f, 65.112544f};

__device__ __forceinline__ float sigf(float x) { return 1.0f / (1.0f + __expf(-x)); }

// ---- WMMA fragment helpers (bf16, 16x16x32) -------------------------------
// B matrix (KxN, 32x16): lanes 0-15 hold K {0..7, 16..23} of column N=lane, lanes 16-31
// hold K {8..15, 24..31} of column N=lane-16  -> two 16B reads per lane.
__device__ __forceinline__ v16bf load_bfrag(const __bf16* W, int stride, int n0,
                                            int kb, int lane) {
  int n = n0 + (lane & 15);
  int k = kb + ((lane >> 4) << 3);           // +0 or +8
  union { uint4 u[2]; v16bf v; } r;
  const uint4* p = (const uint4*)(W + (size_t)n * stride + k);
  r.u[0] = p[0];
  r.u[1] = p[2];
  return r.v;
}
// A matrix (MxK, 16x32): lane m (0-15): K {kb..kb+7, kb+16..kb+23};
// lane m+16: K {kb+8..kb+15, kb+24..kb+31}  -> two 16B LDS reads per lane.
__device__ __forceinline__ v16bf load_afrag(const __bf16* A, int stride,
                                            int kb, int lane) {
  int m = lane & 15;
  int k = kb + ((lane >> 4) << 3);           // +0 or +8
  union { uint4 u[2]; v16bf v; } r;
  r.u[0] = *(const uint4*)(A + (size_t)m * stride + k);
  r.u[1] = *(const uint4*)(A + (size_t)m * stride + k + 16);
  return r.v;
}
__device__ __forceinline__ v8f wmma_bf(v16bf a, v16bf b, v8f c) {
  return __builtin_amdgcn_wmma_f32_16x16x32_bf16(false, a, false, b,
                                                 (short)0, c, false, false);
}

#if HAVE_TDM
// ---- Tensor Data Mover: 2-D tile (global -> LDS), D# per ISA 08 §8 --------
__device__ __forceinline__ void tdm_load_2d(unsigned lds_off, const void* gptr,
                                            unsigned dsz_code,      // 0:1B 1:2B 2:4B
                                            unsigned tile_d0, unsigned tile_d1,
                                            unsigned tensor_d0, unsigned tensor_d1,
                                            unsigned long long stride0) {
  unsigned long long ga = (unsigned long long)(size_t)gptr;
  v4u g0;
  g0.x = 1u;                                          // count=1, user mode
  g0.y = lds_off;                                     // LDS byte address
  g0.z = (unsigned)ga;                                // global addr [31:0]
  g0.w = (unsigned)((ga >> 32) & 0x1FFFFFFu) | (2u << 30);  // addr[56:32] | type=2
  v8i g1;
  g1[0] = (int)(dsz_code << 16);                      // data_size; no mask/pad
  g1[1] = (int)((tensor_d0 & 0xFFFFu) << 16);         // tensor_dim0[15:0]
  g1[2] = (int)(((tensor_d0 >> 16) & 0xFFFFu) | ((tensor_d1 & 0xFFFFu) << 16));
  g1[3] = (int)(((tensor_d1 >> 16) & 0xFFFFu) | ((tile_d0 & 0xFFFFu) << 16));
  g1[4] = (int)(tile_d1 & 0xFFFFu);                   // tile_dim1; tile_dim2=0
  g1[5] = (int)(unsigned)(stride0 & 0xFFFFFFFFu);     // dim0 stride lo
  g1[6] = (int)(unsigned)((stride0 >> 32) & 0xFFFFu); // dim0 stride hi; dim1 stride=0
  g1[7] = 0;
  v4i gz = {0, 0, 0, 0};
  TDM_LOAD(g0, g1, gz, gz);
}
#endif

// ---- grid-wide barrier for the persistent kernel --------------------------
__device__ __forceinline__ void grid_sync(unsigned* cnt, unsigned& epoch) {
  __syncthreads();
  epoch += NWG;
  if (threadIdx.x == 0) {
    __threadfence();
    __hip_atomic_fetch_add(cnt, 1u, __ATOMIC_RELEASE, __HIP_MEMORY_SCOPE_AGENT);
    while (__hip_atomic_load(cnt, __ATOMIC_ACQUIRE, __HIP_MEMORY_SCOPE_AGENT) < epoch)
      __builtin_amdgcn_s_sleep(2);
  }
  __syncthreads();
  __threadfence();
}

// ===========================================================================
// Kernel 1: pack weights to bf16.  Wpack[n][0:512]=W_hh[n][:], [512:768]=W_ih[n][:]
// ===========================================================================
__global__ void k_pack(const float* __restrict__ Wih, const float* __restrict__ Whh,
                       const float* __restrict__ bih, const float* __restrict__ bhh,
                       const float* __restrict__ Wt1,
                       __bf16* __restrict__ Wpack, __bf16* __restrict__ Wt1b,
                       float* __restrict__ bsum) {
  const int total = G4 * KPACK + H_ * 2 * H_ + G4;
  for (int idx = blockIdx.x * blockDim.x + threadIdx.x; idx < total;
       idx += gridDim.x * blockDim.x) {
    if (idx < G4 * KPACK) {
      int n = idx / KPACK, k = idx - n * KPACK;
      float v = (k < H_) ? Whh[n * H_ + k] : Wih[n * I_ + (k - H_)];
      Wpack[idx] = (__bf16)v;
    } else if (idx < G4 * KPACK + H_ * 2 * H_) {
      int j = idx - G4 * KPACK;
      Wt1b[j] = (__bf16)Wt1[j];              // (H x 2H) row-major, direct copy
    } else {
      int n = idx - G4 * KPACK - H_ * 2 * H_;
      bsum[n] = bih[n] + bhh[n];
    }
  }
}

// ===========================================================================
// Kernel 1b: convert x to bf16 once (TDM source; removes per-step converts)
// ===========================================================================
__global__ void k_xbf(const float* __restrict__ x, __bf16* __restrict__ xbf) {
  const int total = B_ * S_ * I_;
  for (int idx = blockIdx.x * blockDim.x + threadIdx.x; idx < total;
       idx += gridDim.x * blockDim.x)
    xbf[idx] = (__bf16)x[idx];
}

// ===========================================================================
// Kernel 2: fold resonance trig into per-t vectors:
//   u_c[t,i] = sum_m Wproj[m,i] cos(g_m t) phi_m ; d_c[t] from b_proj; zeta[t]
// ===========================================================================
__global__ void k_trig(const float* __restrict__ Wproj, const float* __restrict__ bproj,
                       float* __restrict__ uc, float* __restrict__ us,
                       float* __restrict__ dc, float* __restrict__ ds,
                       float* __restrict__ zeta) {
  int idx = blockIdx.x * blockDim.x + threadIdx.x;
  if (idx >= S_ * I_) return;
  int t = idx / I_, i = idx - t * I_;
  float tf = (float)t, ac = 0.f, as = 0.f;
#pragma unroll
  for (int m = 0; m < MM; ++m) {
    float phi = __expf(-0.1f * c_gam[m]);
    float zc  = cosf(c_gam[m] * tf) * phi;
    float zs  = sinf(c_gam[m] * tf) * phi;
    ac += Wproj[m * I_ + i] * zc;
    as += Wproj[(MM + m) * I_ + i] * zs;
  }
  uc[idx] = ac;
  us[idx] = as;
  if (i == 0) {
    float bc = 0.f, bs = 0.f, zw = 0.f;
#pragma unroll
    for (int m = 0; m < MM; ++m) {
      float phi = __expf(-0.1f * c_gam[m]);
      float zc  = cosf(c_gam[m] * tf);
      bc += bproj[m] * zc * phi;
      bs += bproj[MM + m] * sinf(c_gam[m] * tf) * phi;
      zw += phi * zc;
    }
    dc[t] = bc; ds[t] = bs; zeta[t] = zw / (float)MM;
  }
}

// ===========================================================================
// Kernel 3: resonance per (b,t): one wave per row, 256-long dot x 2
// ===========================================================================
__global__ void k_res(const float* __restrict__ x, const float* __restrict__ uc,
                      const float* __restrict__ us, const float* __restrict__ dc,
                      const float* __restrict__ ds, float* __restrict__ res) {
  int wid  = (blockIdx.x * blockDim.x + threadIdx.x) >> 5;
  int lane = threadIdx.x & 31;
  if (wid >= B_ * S_) return;
  int t = wid & (S_ - 1);
  const float* xp = x + (size_t)wid * I_;
  float ca = 0.f, sa = 0.f;
  for (int i = lane; i < I_; i += 32) {
    float xv = xp[i];
    ca += xv * uc[t * I_ + i];
    sa += xv * us[t * I_ + i];
  }
  for (int m = 16; m > 0; m >>= 1) {
    ca += __shfl_xor(ca, m, 32);
    sa += __shfl_xor(sa, m, 32);
  }
  if (lane == 0) {
    ca += dc[t]; sa += ds[t];
    res[wid] = sigf(sqrtf(ca * ca + sa * sa + 1e-8f) - 0.5f);
  }
}

// ===========================================================================
// Kernel 4: zero h buffers, tension accumulators, barrier counter
// ===========================================================================
__global__ void k_init(float* __restrict__ hbuf, float* __restrict__ tens,
                       unsigned* __restrict__ cnt) {
  const int total = 2 * B_ * H_ + 2 * B_ + 1;
  for (int idx = blockIdx.x * blockDim.x + threadIdx.x; idx < total;
       idx += gridDim.x * blockDim.x) {
    if (idx < 2 * B_ * H_)               hbuf[idx] = 0.f;
    else if (idx < 2 * B_ * H_ + 2 * B_) tens[idx - 2 * B_ * H_] = 0.f;
    else                                 *cnt = 0u;
  }
}

// ===========================================================================
// Kernel 5: persistent recurrent kernel.
// block = (tile, chunk): tile = 16 batch rows, chunk = 64 hidden columns.
// wave w owns hidden tile jbase = chunk*64 + w*16 and its 4 gate columns.
// ===========================================================================
extern __shared__ char smem[];
__global__ void __launch_bounds__(TPB)
k_lstm(const __bf16* __restrict__ xbf, const float* __restrict__ res,
       const float* __restrict__ zeta,
       const __bf16* __restrict__ Wpack, const __bf16* __restrict__ Wt1b,
       const float* __restrict__ bsum, const float* __restrict__ bt1,
       const float* __restrict__ wt2, const float* __restrict__ bt2,
       const float* __restrict__ wg, const float* __restrict__ bg,
       float* __restrict__ hbuf, float* __restrict__ tens,
       unsigned* __restrict__ cnt, float* __restrict__ out) {
  // LDS layout (dynamic, 204800 B total)
  __bf16* sAh   = (__bf16*)smem;                                   // 16 x 512 bf16
  __bf16* sAx   = (__bf16*)(smem + 16384);                         // 16 x 256 bf16 (TDM dst)
  __bf16* sComp = (__bf16*)(smem + 16384 + 8192);                  // 16 x 1024
  __bf16* sHist = (__bf16*)(smem + 16384 + 8192 + 32768);          // 5 x 16 x 512
  float*  sHsum = (float*)(smem + 16384 + 8192 + 32768 + 81920);   // 16 x 512
  float*  sHold = (float*)(smem + 16384 + 8192 + 32768 + 81920 + 32768); // 16x512 (TDM dst)

  const int tid   = threadIdx.x;
  const int lane  = tid & 31;
  const int tile  = blockIdx.x >> 3;
  const int chunk = blockIdx.x & 7;
  const int wave  = tid >> 5;
  const int jbase = chunk * JCH + wave * 16;
  const int nlo   = lane & 15;
  const int jcol  = jbase + nlo;                // this lane's hidden column

  // zero LDS history state
  for (int i = tid; i < WIN_ * 16 * H_; i += TPB) sHist[i] = (__bf16)0.f;
  for (int i = tid; i < 16 * H_; i += TPB)        sHsum[i] = 0.f;

  // per-lane constants
  const float bs_i = bsum[jcol],          bs_f = bsum[H_ + jcol];
  const float bs_g = bsum[2 * H_ + jcol], bs_o = bsum[3 * H_ + jcol];
  const float bt1_l = bt1[jcol], wt2_l = wt2[jcol];
  const float bt2_s = bt2[0], wg0 = wg[0], wg1 = wg[1], bg_s = bg[0];

  v8f c_st = {};                 // cell state, register-resident across steps
  unsigned epoch = 0;
  __syncthreads();

  for (int t = 0; t < S_; ++t) {
    const int slot = t % WIN_;
    const float* hb_in  = hbuf + (size_t)(t & 1) * B_ * H_;
    float*       hb_out = hbuf + (size_t)((t + 1) & 1) * B_ * H_;

    // ---- phase 0a: DMA h tile (f32) and x tile (bf16) into LDS -----------
#if HAVE_TDM
    if (wave == 0) {
      // h tile: 16 rows x 512 f32, contiguous 32KB  -> sHold
      tdm_load_2d((unsigned)(size_t)sHold, hb_in + (size_t)tile * 16 * H_,
                  /*dsz=*/2u, /*tile_d0=*/16u * H_, /*tile_d1=*/0u,
                  /*tensor_d0=*/16u * H_, /*tensor_d1=*/0u,
                  /*stride0=*/(unsigned long long)(16 * H_));
      // x tile: 16 rows x 256 bf16, row stride S*I elements -> sAx (contiguous)
      tdm_load_2d((unsigned)(size_t)sAx,
                  xbf + ((size_t)(tile * 16) * S_ + t) * I_,
                  /*dsz=*/1u, /*tile_d0=*/(unsigned)I_, /*tile_d1=*/16u,
                  /*tensor_d0=*/(unsigned)I_, /*tensor_d1=*/16u,
                  /*stride0=*/(unsigned long long)(S_ * I_));
      __builtin_amdgcn_s_wait_tensorcnt(0);
    }
    __syncthreads();
#else
    for (int idx = tid; idx < 16 * H_; idx += TPB)
      sHold[idx] = hb_in[(size_t)tile * 16 * H_ + idx];
    for (int idx = tid; idx < 16 * I_; idx += TPB) {
      int m = idx >> 8, i = idx & (I_ - 1);
      sAx[idx] = xbf[((size_t)(tile * 16 + m) * S_ + t) * I_ + i];
    }
    __syncthreads();
#endif

    // ---- phase 0b: convert h, update history ring / mean -----------------
    for (int idx = tid; idx < 16 * H_; idx += TPB) {
      int m = idx >> 9, j = idx & (H_ - 1);
      float hv = sHold[idx];
      sAh[idx] = (__bf16)hv;
      float oldv = (float)sHist[slot * 16 * H_ + idx];
      float ns   = sHsum[idx] + hv - oldv;        // running history sum
      sHsum[idx] = ns;
      sHist[slot * 16 * H_ + idx] = (__bf16)hv;
      sComp[m * (2 * H_) + j]      = (__bf16)hv;          // h
      sComp[m * (2 * H_) + H_ + j] = (__bf16)(ns * 0.2f); // h_mean
    }
    if (chunk == 0 && tid < 16)                    // reset slot for step t+1
      tens[((t + 1) & 1) * B_ + tile * 16 + tid] = 0.f;
    __syncthreads();

    // ---- phase 1: WMMA GEMMs --------------------------------------------
    v8f acc_i = {}, acc_f = {}, acc_g = {}, acc_o = {};
    for (int kb = 0; kb < H_; kb += 32) {          // h @ W_hh^T part
      v16bf a = load_afrag(sAh, H_, kb, lane);
      acc_i = wmma_bf(a, load_bfrag(Wpack, KPACK, jbase,          kb, lane), acc_i);
      acc_f = wmma_bf(a, load_bfrag(Wpack, KPACK, H_ + jbase,     kb, lane), acc_f);
      acc_g = wmma_bf(a, load_bfrag(Wpack, KPACK, 2 * H_ + jbase, kb, lane), acc_g);
      acc_o = wmma_bf(a, load_bfrag(Wpack, KPACK, 3 * H_ + jbase, kb, lane), acc_o);
    }
    for (int kb = 0; kb < I_; kb += 32) {          // x @ W_ih^T part
      v16bf a = load_afrag(sAx, I_, kb, lane);
      acc_i = wmma_bf(a, load_bfrag(Wpack, KPACK, jbase,          H_ + kb, lane), acc_i);
      acc_f = wmma_bf(a, load_bfrag(Wpack, KPACK, H_ + jbase,     H_ + kb, lane), acc_f);
      acc_g = wmma_bf(a, load_bfrag(Wpack, KPACK, 2 * H_ + jbase, H_ + kb, lane), acc_g);
      acc_o = wmma_bf(a, load_bfrag(Wpack, KPACK, 3 * H_ + jbase, H_ + kb, lane), acc_o);
    }
    v8f acc_t = {};
    for (int kb = 0; kb < 2 * H_; kb += 32) {      // tension layer-1
      v16bf a = load_afrag(sComp, 2 * H_, kb, lane);
      acc_t = wmma_bf(a, load_bfrag(Wt1b, 2 * H_, jbase, kb, lane), acc_t);
    }
    // tension partial: relu -> dot(W_t2 chunk) -> per-row atomic partial sum
#pragma unroll
    for (int r = 0; r < 8; ++r) {
      float v = fmaxf(acc_t[r] + bt1_l, 0.f) * wt2_l;
      for (int msk = 8; msk > 0; msk >>= 1) v += __shfl_xor(v, msk, 16);
      if (nlo == 0) {
        int mrow = r + ((lane >> 4) << 3);
        __hip_atomic_fetch_add(&tens[(t & 1) * B_ + tile * 16 + mrow], v,
                               __ATOMIC_RELAXED, __HIP_MEMORY_SCOPE_AGENT);
      }
    }
    grid_sync(cnt, epoch);                         // barrier A

    // ---- phase 2: cell update + gated output ----------------------------
    const float zt = zeta[t];
#pragma unroll
    for (int r = 0; r < 8; ++r) {
      int mrow = r + ((lane >> 4) << 3);
      int row  = tile * 16 + mrow;
      float gi = sigf(acc_i[r] + bs_i);
      float gf = sigf(acc_f[r] + bs_f);
      float gg = tanhf(acc_g[r] + bs_g);
      float go = sigf(acc_o[r] + bs_o);
      float cn = gf * c_st[r] + gi * gg;
      c_st[r] = cn;
      float hn = go * tanhf(cn);
      float tsum = __hip_atomic_load(&tens[(t & 1) * B_ + tile * 16 + mrow],
                                     __ATOMIC_RELAXED, __HIP_MEMORY_SCOPE_AGENT);
      float tension = (t < 2) ? 0.f : sigf(tsum + bt2_s);
      float rv   = res[(size_t)row * S_ + t];
      float gate = sigf(wg0 * rv + wg1 * tension + bg_s);
      float ho   = hn + gate * (zt * sHold[mrow * H_ + jcol]);
      hb_out[row * H_ + jcol] = ho;
      out[((size_t)row * S_ + t) * H_ + jcol] = ho;
      if (t == S_ - 1) {
        out[(size_t)B_ * S_ * H_ + (size_t)row * H_ + jcol] = ho;                   // h_n
        out[(size_t)B_ * S_ * H_ + (size_t)B_ * H_ + (size_t)row * H_ + jcol] = cn; // c_n
      }
    }
    grid_sync(cnt, epoch);                         // barrier B
  }
}

// ===========================================================================
extern "C" void kernel_launch(void* const* d_in, const int* in_sizes, int n_in,
                              void* d_out, int out_size, void* d_ws, size_t ws_size,
                              hipStream_t stream) {
  (void)in_sizes; (void)n_in; (void)out_size; (void)ws_size;
  const float* x     = (const float*)d_in[0];
  const float* Wih   = (const float*)d_in[1];
  const float* Whh   = (const float*)d_in[2];
  const float* bih   = (const float*)d_in[3];
  const float* bhh   = (const float*)d_in[4];
  const float* Wproj = (const float*)d_in[5];
  const float* bproj = (const float*)d_in[6];
  const float* Wt1   = (const float*)d_in[7];
  const float* bt1   = (const float*)d_in[8];
  const float* wt2   = (const float*)d_in[9];
  const float* bt2   = (const float*)d_in[10];
  const float* wg    = (const float*)d_in[11];
  const float* bg    = (const float*)d_in[12];
  float* out = (float*)d_out;

  char* ws = (char*)d_ws;
  size_t off = 0;
  auto carve = [&](size_t bytes) {
    void* p = ws + off;
    off = (off + bytes + 255) & ~(size_t)255;
    return p;
  };
  __bf16*   wpack = (__bf16*)carve((size_t)G4 * KPACK * 2);
  __bf16*   wt1b  = (__bf16*)carve((size_t)H_ * 2 * H_ * 2);
  __bf16*   xbf   = (__bf16*)carve((size_t)B_ * S_ * I_ * 2);
  float*    bsum  = (float*)carve(G4 * 4);
  float*    uc    = (float*)carve((size_t)S_ * I_ * 4);
  float*    us    = (float*)carve((size_t)S_ * I_ * 4);
  float*    dc    = (float*)carve(S_ * 4);
  float*    ds    = (float*)carve(S_ * 4);
  float*    zeta  = (float*)carve(S_ * 4);
  float*    res   = (float*)carve((size_t)B_ * S_ * 4);
  float*    hbuf  = (float*)carve((size_t)2 * B_ * H_ * 4);
  float*    tens  = (float*)carve(2 * B_ * 4);
  unsigned* cnt   = (unsigned*)carve(256);

  k_pack<<<2048, 256, 0, stream>>>(Wih, Whh, bih, bhh, Wt1, wpack, wt1b, bsum);
  k_xbf<<<8192, 256, 0, stream>>>(x, xbf);
  k_trig<<<(S_ * I_ + 255) / 256, 256, 0, stream>>>(Wproj, bproj, uc, us, dc, ds, zeta);
  k_res<<<(B_ * S_) / 8, 256, 0, stream>>>(x, uc, us, dc, ds, res);
  k_init<<<1024, 256, 0, stream>>>(hbuf, tens, cnt);

  const size_t smem_bytes = 16384 + 8192 + 32768 + 81920 + 32768 + 32768; // 204800 B
  k_lstm<<<NWG, TPB, smem_bytes, stream>>>(xbf, res, zeta, wpack, wt1b, bsum, bt1,
                                           wt2, bt2, wg, bg, hbuf, tens, cnt, out);
}
